// MoEConnectionProcessor_28810640622311
// MI455X (gfx1250) — matrix operand order repaired and checked by reading, not compiled, not run on hardware
//
#include <hip/hip_runtime.h>
#include <hip/hip_bf16.h>
#include <math.h>
#include <stdint.h>

typedef _Float16 f16;
typedef __attribute__((ext_vector_type(16))) _Float16 v16h;
typedef __attribute__((ext_vector_type(8)))  _Float16 v8h;
typedef __attribute__((ext_vector_type(4)))  _Float16 v4h;
typedef __attribute__((ext_vector_type(8)))  float    v8f;
typedef __attribute__((ext_vector_type(4)))  float    v4f;

#define NCELLS 19683
#define KNBR   26
#define DIM    128
#define MPAD   19696      // 1231 * 16
#define MTILES 1231
#define LSIDE  27
#define DT     0.1f
#define CSTRIDE 896       // 7 * 128 fused output columns
// column offsets in the fused stage-1 GEMM output
#define OFF_P   0         // states @ Wm_top
#define OFF_Q   128       // states @ Wm_bot   (gathered per neighbor)
#define OFF_L1  256       // states @ Wl_top
#define OFF_L2  384       // states @ Wl_bot   (gathered)
#define OFF_U   512       // states @ Wu
#define OFF_S   640       // states @ Wc_bot   (gathered)
#define OFF_C1  768       // states @ Wc_top

// ---------------- small vector helpers ----------------
__device__ __forceinline__ v4f ld4(const float* p) { return *(const v4f*)p; }
__device__ __forceinline__ void st4(float* p, v4f v) { *(v4f*)p = v; }
__device__ __forceinline__ v4f tanh4(v4f a) {
  v4f r; r[0]=tanhf(a[0]); r[1]=tanhf(a[1]); r[2]=tanhf(a[2]); r[3]=tanhf(a[3]); return r;
}
__device__ __forceinline__ v4f relu4(v4f a) {
  v4f r; r[0]=fmaxf(a[0],0.f); r[1]=fmaxf(a[1],0.f); r[2]=fmaxf(a[2],0.f); r[3]=fmaxf(a[3],0.f); return r;
}
__device__ __forceinline__ v4f mul4s(v4f a, float s) {
  v4f r; r[0]=a[0]*s; r[1]=a[1]*s; r[2]=a[2]*s; r[3]=a[3]*s; return r;
}
__device__ __forceinline__ void st4h(f16* p, v4f a) {
  v4h h; h[0]=(f16)a[0]; h[1]=(f16)a[1]; h[2]=(f16)a[2]; h[3]=(f16)a[3]; *(v4h*)p = h;
}

// ---------------- weight prepack into WMMA-B layout ----------------
// B tile (32x16 f16) per lane: column N = g*16 + (lane&15); half i -> K = kt*32 + (lane>>4)*16 + i
// 64 global ntile groups: m = g/8 selects source matrix block.
__global__ void k_pack_w(const float* __restrict__ Wm, const float* __restrict__ Wl,
                         const float* __restrict__ Wu, const float* __restrict__ Wc,
                         const float* __restrict__ Wo, f16* __restrict__ Bp) {
  int t = blockIdx.x * blockDim.x + threadIdx.x;
  if (t >= 64*4*32) return;
  int lane = t & 31;
  int kt   = (t >> 5) & 3;
  int g    = t >> 7;
  int m    = g >> 3;
  int n    = (g & 7) * 16 + (lane & 15);
  int kbase = kt*32 + (lane >> 4)*16;
  f16* dst = Bp + ((size_t)(g*4 + kt)*32 + lane)*16;
  #pragma unroll
  for (int i = 0; i < 16; ++i) {
    int K = kbase + i;
    float v;
    switch (m) {
      case 0: v = Wm[(size_t)K*DIM + n];        break;  // Wm_top
      case 1: v = Wm[(size_t)(K+DIM)*DIM + n];  break;  // Wm_bot
      case 2: v = Wl[(size_t)K*DIM + n];        break;  // Wl_top
      case 3: v = Wl[(size_t)(K+DIM)*DIM + n];  break;  // Wl_bot
      case 4: v = Wu[(size_t)K*DIM + n];        break;  // Wu
      case 5: v = Wc[(size_t)(K+DIM)*DIM + n];  break;  // Wc_bot
      case 6: v = Wc[(size_t)K*DIM + n];        break;  // Wc_top
      default: v = Wo[(size_t)K*DIM + n];       break;  // Wo
    }
    dst[i] = (f16)v;
  }
}

__global__ void k_pack_states(const float* __restrict__ s, f16* __restrict__ o) {
  int t = blockIdx.x * blockDim.x + threadIdx.x;
  if (t >= MPAD*DIM) return;
  int row = t / DIM;
  o[t] = (row < NCELLS) ? (f16)s[t] : (f16)0.f;
}

__global__ void k_zero_pad(f16* __restrict__ Fh, f16* __restrict__ Xh) {
  int t = blockIdx.x * blockDim.x + threadIdx.x;
  if (t >= (MPAD-NCELLS)*DIM) return;
  size_t i = (size_t)NCELLS*DIM + t;
  Fh[i] = (f16)0.f; Xh[i] = (f16)0.f;
}

// ---------------- gating pre-projections (tiny) ----------------
__global__ void k_gatespre(const float* __restrict__ states, const float* __restrict__ Wg,
                           const float* __restrict__ bg,
                           float* __restrict__ G1, float* __restrict__ G2) {
  int t = blockIdx.x * blockDim.x + threadIdx.x;
  if (t >= NCELLS*3) return;
  int r = t / 3, j = t % 3;
  const float* s = states + (size_t)r * DIM;
  float a = 0.f, b = 0.f;
  for (int d = 0; d < DIM; ++d) {
    float v = s[d];
    a += v * Wg[(size_t)d*3 + j];
    b += v * Wg[(size_t)(d+DIM)*3 + j];
  }
  G1[t] = a + bg[j];
  G2[t] = b;
}

// ---------------- WMMA GEMM: C[M x ntiles*16] = A[M x 128](f16) @ Bpacked ----------------
// one wave computes a 16-row M tile x 4 consecutive 16-col N tiles, K=128 in 4 chunks of 32
__global__ void k_gemm(const f16* __restrict__ A, const f16* __restrict__ Bp,
                       float* __restrict__ C, int ntiles, int cstride, int mvalid) {
  int mtile = blockIdx.x;
  int lane  = threadIdx.x;               // 0..31
  int g0    = (blockIdx.y * 4 + threadIdx.y) * 4;
  if (g0 >= ntiles) return;
  int laneLo = lane & 15, laneHi = lane >> 4;
  const f16* arow = A + (size_t)(mtile*16 + laneLo) * DIM;
  v8f acc[4] = {};
  #pragma unroll
  for (int kt = 0; kt < 4; ++kt) {
    int aoff = kt*32 + laneHi*16;
    v8h alo = *(const v8h*)(arow + aoff);
    v8h ahi = *(const v8h*)(arow + aoff + 8);
    v16h av = __builtin_shufflevector(alo, ahi, 0,1,2,3,4,5,6,7,8,9,10,11,12,13,14,15);
    #pragma unroll
    for (int t = 0; t < 4; ++t) {
      int g = g0 + t;
      if (g >= ntiles) break;            // wave-uniform
      const f16* bptr = Bp + ((size_t)(g*4 + kt)*32 + lane)*16;
      v8h blo = *(const v8h*)bptr;
      v8h bhi = *(const v8h*)(bptr + 8);
      v16h bv = __builtin_shufflevector(blo, bhi, 0,1,2,3,4,5,6,7,8,9,10,11,12,13,14,15);
      acc[t] = __builtin_amdgcn_wmma_f32_16x16x32_f16(false, av, false, bv,
                                                      (short)0, acc[t], false, false);
    }
  }
  #pragma unroll
  for (int t = 0; t < 4; ++t) {
    int g = g0 + t;
    if (g >= ntiles) break;
    int col = g*16 + laneLo;
    #pragma unroll
    for (int v = 0; v < 8; ++v) {
      int row = mtile*16 + laneHi*8 + v;
      if (row < mvalid) C[(size_t)row*cstride + col] = acc[t][v];
    }
  }
}

// ---------------- fusion A: gathers, masks, local expert, CNF step 1, gating ----------------
__global__ void k_fusionA(const float* __restrict__ states, const int* __restrict__ cellIdx,
                          const int* __restrict__ nbr, const float* __restrict__ Call,
                          const float* __restrict__ G1, const float* __restrict__ G2,
                          const float* __restrict__ bl, const float* __restrict__ bm,
                          const float* __restrict__ bc,
                          f16* __restrict__ Fh, f16* __restrict__ Xh,
                          float* __restrict__ X, float* __restrict__ Sagg,
                          float* __restrict__ gates, float* __restrict__ out) {
  int wave = threadIdx.x >> 5, lane = threadIdx.x & 31;
  int b = blockIdx.x * 8 + wave;
  if (b >= NCELLS) return;
  int ci = cellIdx[b];
  int c0 = lane * 4;
  const float* crow = Call + (size_t)ci * CSTRIDE;
  v4f cur = ld4(states + (size_t)ci*DIM + c0);
  v4f pb  = ld4(crow + OFF_P + c0) + ld4(bm + c0);   // P[b] + bm hoisted out of relu sum
  int bx = ci % LSIDE, by = (ci / LSIDE) % LSIDE, bz = ci / (LSIDE*LSIDE);
  v4f sumL = {0.f,0.f,0.f,0.f}, sumF = {0.f,0.f,0.f,0.f}, sumS = {0.f,0.f,0.f,0.f};
  float g2sum = 0.f;
  int nl = 0, nf = 0, nd = 0;
  for (int k = 0; k < KNBR; ++k) {
    int n = nbr[(size_t)b*KNBR + k];
    int nx = n % LSIDE, ny = (n / LSIDE) % LSIDE, nz = n / (LSIDE*LSIDE);
    int dx = bx-nx, dy = by-ny, dz = bz-nz;
    int d2 = dx*dx + dy*dy + dz*dz;                  // exact integer distance^2
    const float* nrow = Call + (size_t)n * CSTRIDE;
    if (d2 <= 3)        { nl++; sumL += ld4(nrow + OFF_L2 + c0); }          // dist <= 1.8
    else if (d2 >= 13)  { nd++; sumS += ld4(nrow + OFF_S  + c0); }          // dist > 3.5
    else                { nf++; sumF += relu4(pb + ld4(nrow + OFF_Q + c0)); }
    if (lane < 3) g2sum += G2[(size_t)n*3 + lane];
  }
  // local expert
  v4f localAgg = mul4s(sumL, 1.f / (float)(nl > 0 ? nl : 1));
  v4f lo = tanh4(ld4(crow + OFF_L1 + c0) + localAgg + ld4(bl + c0));
  // functional aggregate -> f16 for Wo GEMM
  st4h(Fh + (size_t)b*DIM + c0, mul4s(sumF, 1.f / (float)(nf > 0 ? nf : 1)));
  // distant aggregate (constant across CNF steps)
  v4f S4 = mul4s(sumS, 1.f / (float)(nd > 0 ? nd : 1));
  st4(Sagg + (size_t)b*DIM + c0, S4);
  // CNF Euler step 1 (x0 = cur, cur@Wc_top precomputed as C1)
  v4f bc4 = ld4(bc + c0);
  v4f x1 = cur + mul4s(tanh4(ld4(crow + OFF_C1 + c0) + S4 + bc4), DT);
  st4(X + (size_t)b*DIM + c0, x1);
  st4h(Xh + (size_t)b*DIM + c0, x1);
  // gating softmax
  float lv = (lane < 3) ? (G1[(size_t)ci*3 + lane] + g2sum * (1.f/26.f)) : 0.f;
  float l0 = __shfl(lv, 0), l1 = __shfl(lv, 1), l2 = __shfl(lv, 2);
  float mx = fmaxf(l0, fmaxf(l1, l2));
  float e0 = __expf(l0-mx), e1 = __expf(l1-mx), e2 = __expf(l2-mx);
  float inv = 1.f / (e0 + e1 + e2);
  float gv0 = e0*inv, gv1 = e1*inv, gv2 = e2*inv;
  if (lane == 0) { gates[(size_t)b*3] = gv0; gates[(size_t)b*3+1] = gv1; gates[(size_t)b*3+2] = gv2; }
  // partial output: cur + g0 * local_out
  st4(out + (size_t)b*DIM + c0, cur + mul4s(lo, gv0));
}

// ---------------- fusion B: functional expert output, CNF step 2 ----------------
__global__ void k_fusionB(const int* __restrict__ cellIdx, const float* __restrict__ Call,
                          const float* __restrict__ FO, const float* __restrict__ CX,
                          const float* __restrict__ Sagg, const float* __restrict__ gates,
                          const float* __restrict__ bo, const float* __restrict__ bc,
                          float* __restrict__ X, f16* __restrict__ Xh, float* __restrict__ out) {
  int wave = threadIdx.x >> 5, lane = threadIdx.x & 31;
  int b = blockIdx.x * 8 + wave;
  if (b >= NCELLS) return;
  int ci = cellIdx[b];
  int c0 = lane * 4;
  size_t oi = (size_t)b*DIM + c0;
  v4f fo = tanh4(ld4(FO + oi) + ld4(Call + (size_t)ci*CSTRIDE + OFF_U + c0) + ld4(bo + c0));
  float gv1 = gates[(size_t)b*3 + 1];
  st4(out + oi, ld4(out + oi) + mul4s(fo, gv1));
  v4f x2 = ld4(X + oi) + mul4s(tanh4(ld4(CX + oi) + ld4(Sagg + oi) + ld4(bc + c0)), DT);
  st4(X + oi, x2);
  st4h(Xh + oi, x2);
}

// ---------------- fusion C: CNF step 3, distant expert, final combine ----------------
__global__ void k_fusionC(const float* __restrict__ states, const int* __restrict__ cellIdx,
                          const float* __restrict__ Sagg, const float* __restrict__ gates,
                          const float* __restrict__ bc, const float* __restrict__ CX,
                          const float* __restrict__ X, float* __restrict__ out) {
  int wave = threadIdx.x >> 5, lane = threadIdx.x & 31;
  int b = blockIdx.x * 8 + wave;
  if (b >= NCELLS) return;
  int ci = cellIdx[b];
  int c0 = lane * 4;
  size_t oi = (size_t)b*DIM + c0;
  v4f x3 = ld4(X + oi) + mul4s(tanh4(ld4(CX + oi) + ld4(Sagg + oi) + ld4(bc + c0)), DT);
  v4f cur = ld4(states + (size_t)ci*DIM + c0);
  float gv2 = gates[(size_t)b*3 + 2];
  st4(out + oi, ld4(out + oi) + mul4s(x3 - cur, gv2));
}

extern "C" void kernel_launch(void* const* d_in, const int* in_sizes, int n_in,
                              void* d_out, int out_size, void* d_ws, size_t ws_size,
                              hipStream_t stream) {
  const float* states = (const float*)d_in[0];
  const float* Wg     = (const float*)d_in[1];
  const float* bg     = (const float*)d_in[2];
  const float* Wl     = (const float*)d_in[3];
  const float* bl     = (const float*)d_in[4];
  const float* Wm     = (const float*)d_in[5];
  const float* bm     = (const float*)d_in[6];
  const float* Wo     = (const float*)d_in[7];
  const float* Wu     = (const float*)d_in[8];
  const float* bo     = (const float*)d_in[9];
  const float* Wc     = (const float*)d_in[10];
  const float* bc     = (const float*)d_in[11];
  const int* cellIdx  = (const int*)d_in[12];
  const int* nbr      = (const int*)d_in[13];
  float* out = (float*)d_out;
  (void)in_sizes; (void)n_in; (void)out_size; (void)ws_size;

  uintptr_t base = (uintptr_t)d_ws;
  auto carve = [&](size_t bytes) -> void* {
    uintptr_t p = (base + 255) & ~(uintptr_t)255;
    base = p + bytes;
    return (void*)p;
  };
  f16*   statesH = (f16*)  carve((size_t)MPAD*DIM*2);
  f16*   Bp      = (f16*)  carve((size_t)64*4*32*16*2);
  float* Call    = (float*)carve((size_t)NCELLS*CSTRIDE*4);
  float* G1      = (float*)carve((size_t)NCELLS*3*4);
  float* G2      = (float*)carve((size_t)NCELLS*3*4);
  f16*   Fh      = (f16*)  carve((size_t)MPAD*DIM*2);
  f16*   Xh      = (f16*)  carve((size_t)MPAD*DIM*2);
  float* X       = (float*)carve((size_t)NCELLS*DIM*4);
  float* Sg      = (float*)carve((size_t)NCELLS*DIM*4);
  float* FO      = (float*)carve((size_t)NCELLS*DIM*4);
  float* CX      = (float*)carve((size_t)NCELLS*DIM*4);
  float* gates   = (float*)carve((size_t)NCELLS*3*4);

  k_pack_w<<<(64*4*32 + 255)/256, 256, 0, stream>>>(Wm, Wl, Wu, Wc, Wo, Bp);
  k_pack_states<<<(MPAD*DIM + 255)/256, 256, 0, stream>>>(states, statesH);
  k_zero_pad<<<((MPAD-NCELLS)*DIM + 255)/256, 256, 0, stream>>>(Fh, Xh);
  k_gatespre<<<(NCELLS*3 + 255)/256, 256, 0, stream>>>(states, Wg, bg, G1, G2);

  dim3 gblk(32, 4);
  // fused stage-1 GEMM: states @ [Wm_top|Wm_bot|Wl_top|Wl_bot|Wu|Wc_bot|Wc_top]
  k_gemm<<<dim3(MTILES, 4), gblk, 0, stream>>>(statesH, Bp, Call, 56, CSTRIDE, NCELLS);

  int fblocks = (NCELLS + 7) / 8;
  k_fusionA<<<fblocks, 256, 0, stream>>>(states, cellIdx, nbr, Call, G1, G2,
                                         bl, bm, bc, Fh, Xh, X, Sg, gates, out);
  // func_agg @ Wo  (Wo packed at group 56)
  k_gemm<<<dim3(MTILES, 1), gblk, 0, stream>>>(Fh, Bp + (size_t)56*2048, FO, 8, DIM, NCELLS);
  // x1 @ Wc_top    (Wc_top packed at group 48)
  k_gemm<<<dim3(MTILES, 1), gblk, 0, stream>>>(Xh, Bp + (size_t)48*2048, CX, 8, DIM, NCELLS);
  k_fusionB<<<fblocks, 256, 0, stream>>>(cellIdx, Call, FO, CX, Sg, gates, bo, bc, X, Xh, out);
  // x2 @ Wc_top
  k_gemm<<<dim3(MTILES, 1), gblk, 0, stream>>>(Xh, Bp + (size_t)48*2048, CX, 8, DIM, NCELLS);
  k_fusionC<<<fblocks, 256, 0, stream>>>(states, cellIdx, Sg, gates, bc, CX, X, out);
}
